// MultiHeadAttention_7928509628886
// MI455X (gfx1250) — compile-verified, matrix operand order checked by reading, not run on hardware
//
#include <hip/hip_runtime.h>
#include <hip/hip_bf16.h>

#define D_MODEL   1024
#define NUM_HEADS 16
#define D_HEAD    64

typedef __attribute__((ext_vector_type(16))) __bf16 v16bf;
typedef __attribute__((ext_vector_type(8)))  float  v8f;

#ifndef __has_builtin
#define __has_builtin(x) 0
#endif

// ---- CDNA5 async global->LDS staging (guarded: falls back to plain copies) --
#if __has_builtin(__builtin_amdgcn_global_load_async_to_lds_b128)
#define HAVE_ASYNC_COPY 1
#define AS1 __attribute__((address_space(1)))
#define AS3 __attribute__((address_space(3)))
typedef __attribute__((__vector_size__(4 * sizeof(int)))) int v4i;
__device__ __forceinline__ void async_copy_b128(const void* g, void* l) {
  __builtin_amdgcn_global_load_async_to_lds_b128((AS1 v4i*)g, (AS3 v4i*)l, 0, 0);
}
__device__ __forceinline__ void async_wait() {
#if __has_builtin(__builtin_amdgcn_s_wait_asynccnt)
  __builtin_amdgcn_s_wait_asynccnt(0);
#else
  asm volatile("s_wait_asynccnt 0x0" ::: "memory");
#endif
}
#else
#define HAVE_ASYNC_COPY 0
#endif

union Frag16 {
  v16bf v;
  unsigned short u[16];
  uint4 q[2];
};

__device__ __forceinline__ unsigned short f2bf(float f) {
  unsigned int u = __float_as_uint(f);
  unsigned int r = u + 0x7FFFu + ((u >> 16) & 1u);   // round-to-nearest-even
  return (unsigned short)(r >> 16);
}
__device__ __forceinline__ float bf2f(unsigned short u) {
  return __uint_as_float((unsigned int)u << 16);
}

__device__ __forceinline__ v8f wmma_bf16(const Frag16& a, const Frag16& b, v8f c) {
  return __builtin_amdgcn_wmma_f32_16x16x32_bf16(false, a.v, false, b.v,
                                                 (short)0, c, false, false);
}

// 16x32 bf16 fragment from LDS tile stored row-major [row][K] (stride elems).
// A-layout: lane L holds row (L&15); K: base..base+7 and 16+base..16+base+7,
// base=(L&16)?8:0 -> two contiguous ds_load_b128. B-fragments: same pattern
// over the [N][K]-stored matrix.
__device__ __forceinline__ Frag16 load_frag(const unsigned short* lds,
                                            int rowBase, int stride, int kOfs, int lane) {
  Frag16 f;
  int row  = rowBase + (lane & 15);
  int base = (lane & 16) ? 8 : 0;
  const unsigned short* p = lds + row * stride + kOfs + base;
  f.q[0] = *(const uint4*)(p);
  f.q[1] = *(const uint4*)(p + 16);
  return f;
}

// ---------------------------------------------------------------------------
// GEMM: C[M,N] = A[M,K] * W[K,N] + bias.
//   A_BF16:  A operand is bf16 (b128 LDS copy / async DMA, no VALU conversion)
//   OUT_SCT: write bf16 scattered to [B,H,S,Dh]; else fp32 row-major
// Block 64x64, BK=64 (2 WMMA k-steps per stage -> 8 WMMAs between barriers),
// 128 threads (4 waves), wave = 2x2 WMMA tiles.
// ---------------------------------------------------------------------------
template<int A_BF16, int OUT_SCT>
__global__ __launch_bounds__(128) void gemm_wmma(
    const void* __restrict__ Ain, const float* __restrict__ W,
    const float* __restrict__ bias, void* __restrict__ Cout,
    int M, int N, int K, int S)
{
  __shared__ __align__(16) unsigned short sA[64 * 64];   // [m][k]
  __shared__ __align__(16) unsigned short sB[64 * 64];   // [n][k] (W transposed)

  const int tid  = threadIdx.x;
  const int lane = tid & 31;
  const int wave = tid >> 5;
  const int bm = blockIdx.x * 64;
  const int bn = blockIdx.y * 64;
  const int wm = (wave >> 1) * 32;
  const int wn = (wave & 1) * 32;

  v8f acc[2][2] = {};

  for (int k0 = 0; k0 < K; k0 += 64) {
    { // stage A tile 64x64 (each thread: 32 contiguous elements)
      int row = tid >> 1;
      int kof = (tid & 1) * 32;
      if (A_BF16) {
        const uint4* g = (const uint4*)((const unsigned short*)Ain +
                                        (size_t)(bm + row) * K + k0 + kof);
        uint4* d = (uint4*)&sA[row * 64 + kof];
#if HAVE_ASYNC_COPY
        #pragma unroll
        for (int i = 0; i < 4; ++i) async_copy_b128(g + i, d + i);
#else
        #pragma unroll
        for (int i = 0; i < 4; ++i) d[i] = g[i];
#endif
      } else {
        const float4* g = (const float4*)((const float*)Ain +
                                          (size_t)(bm + row) * K + k0 + kof);
        unsigned short* d = &sA[row * 64 + kof];
        #pragma unroll
        for (int i = 0; i < 8; ++i) {
          float4 t = g[i];
          d[4*i+0] = f2bf(t.x); d[4*i+1] = f2bf(t.y);
          d[4*i+2] = f2bf(t.z); d[4*i+3] = f2bf(t.w);
        }
      }
    }
    { // stage W tile (64k x 64n) transposed into [n][k]
      int kr  = tid >> 1;
      int nof = (tid & 1) * 32;
      const float4* g = (const float4*)(W + (size_t)(k0 + kr) * N + bn + nof);
      #pragma unroll
      for (int i = 0; i < 8; ++i) {
        float4 t = g[i];
        sB[(nof + 4*i + 0) * 64 + kr] = f2bf(t.x);
        sB[(nof + 4*i + 1) * 64 + kr] = f2bf(t.y);
        sB[(nof + 4*i + 2) * 64 + kr] = f2bf(t.z);
        sB[(nof + 4*i + 3) * 64 + kr] = f2bf(t.w);
      }
    }
#if HAVE_ASYNC_COPY
    if (A_BF16) async_wait();
#endif
    __syncthreads();

    #pragma unroll
    for (int ks = 0; ks < 2; ++ks)
    #pragma unroll
    for (int mi = 0; mi < 2; ++mi) {
      Frag16 af = load_frag(sA, wm + mi * 16, 64, ks * 32, lane);
      #pragma unroll
      for (int ni = 0; ni < 2; ++ni) {
        Frag16 bf = load_frag(sB, wn + ni * 16, 64, ks * 32, lane);
        acc[mi][ni] = wmma_bf16(af, bf, acc[mi][ni]);
      }
    }
    __syncthreads();
  }

  // Epilogue. D-layout: lane holds col (lane&15); VGPR r -> row r + 8*(lane>=16).
  const int rofs = (lane & 16) ? 8 : 0;
  #pragma unroll
  for (int mi = 0; mi < 2; ++mi)
  #pragma unroll
  for (int ni = 0; ni < 2; ++ni) {
    int n = bn + wn + ni * 16 + (lane & 15);
    float bv = bias[n];
    #pragma unroll
    for (int r = 0; r < 8; ++r) {
      int m = bm + wm + mi * 16 + r + rofs;
      float val = acc[mi][ni][r] + bv;
      if (OUT_SCT) {
        int bb = m / S, ss = m % S;
        int hh = n >> 6, dd = n & 63;
        ((unsigned short*)Cout)[(((size_t)bb * NUM_HEADS + hh) * S + ss) * D_HEAD + dd]
            = f2bf(val);
      } else {
        ((float*)Cout)[(size_t)m * N + n] = val;
      }
    }
  }
}

// ---------------------------------------------------------------------------
// RoPE applied in-place to bf16 Q and K workspace, layout [B,H,S,Dh].
// ---------------------------------------------------------------------------
__global__ void rope_kernel(unsigned short* __restrict__ Q,
                            unsigned short* __restrict__ K,
                            long long npair, int S)
{
  long long idx = (long long)blockIdx.x * blockDim.x + threadIdx.x;
  if (idx >= npair) return;
  int j = (int)(idx & 31);            // rotation pair within head dim
  long long rem = idx >> 5;           // (b*H + h)*S + s
  int s = (int)(rem % S);
  float inv_freq = __expf(-(float)(2 * j) * (9.210340372f / 64.0f)); // 10000^(-2j/64)
  float ang = (float)s * inv_freq;
  float sn, cs;
  __sincosf(ang, &sn, &cs);
  long long base = rem * D_HEAD + 2 * j;
  float a0 = bf2f(Q[base]), a1 = bf2f(Q[base + 1]);
  Q[base]     = f2bf(a0 * cs - a1 * sn);
  Q[base + 1] = f2bf(a1 * cs + a0 * sn);
  float b0 = bf2f(K[base]), b1 = bf2f(K[base + 1]);
  K[base]     = f2bf(b0 * cs - b1 * sn);
  K[base + 1] = f2bf(b1 * cs + b0 * sn);
}

// ---------------------------------------------------------------------------
// Flash attention with fused relative-position bias gather, bf16 Q/K/V/O.
// Grid: (S/64, B*H). Block: 128 threads = 4 waves, each wave owns 16 q rows.
// ---------------------------------------------------------------------------
__global__ __launch_bounds__(128) void attn_wmma(
    const unsigned short* __restrict__ Q, const unsigned short* __restrict__ K,
    const unsigned short* __restrict__ V, const int* __restrict__ dist,
    const float* __restrict__ bias_table, unsigned short* __restrict__ O,
    int S)
{
  __shared__ __align__(16) unsigned short sQ[64 * 64];     // [q][d]
  __shared__ __align__(16) unsigned short sK[64 * 64];     // [k][d]
  __shared__ __align__(16) unsigned short sV[64 * 64];     // [d][k] (transposed)
  __shared__ __align__(16) unsigned short sP[4][16 * 64];  // per-wave P tile

  const int tid  = threadIdx.x;
  const int lane = tid & 31;
  const int wave = tid >> 5;
  const int bh = blockIdx.y;
  const int b  = bh >> 4;
  const int h  = bh & 15;
  const int q0 = blockIdx.x * 64;

  const unsigned short* Qp = Q + ((size_t)bh * S + q0) * D_HEAD;
  const unsigned short* Kb = K + (size_t)bh * S * D_HEAD;
  const unsigned short* Vb = V + (size_t)bh * S * D_HEAD;
  const int* distb = dist + (size_t)b * S * S;

  { // stage Q tile once (64x64 bf16)
    int row = tid >> 1;
    int dof = (tid & 1) * 32;
    const uint4* g = (const uint4*)(Qp + row * 64 + dof);
    uint4* d = (uint4*)&sQ[row * 64 + dof];
#if HAVE_ASYNC_COPY
    #pragma unroll
    for (int i = 0; i < 4; ++i) async_copy_b128(g + i, d + i);
#else
    #pragma unroll
    for (int i = 0; i < 4; ++i) d[i] = g[i];
#endif
    // prefetch first dist block rows while Q stages
    __builtin_prefetch(distb + (size_t)(q0 + (tid & 63)) * S, 0, 1);
#if HAVE_ASYNC_COPY
    async_wait();
#endif
  }
  __syncthreads();

  const int qw   = wave * 16;
  const int rofs = (lane & 16) ? 8 : 0;
  v8f o[4] = {};
  float mrow[8], lrow[8];
  #pragma unroll
  for (int r = 0; r < 8; ++r) { mrow[r] = -1e30f; lrow[r] = 0.f; }

  for (int k0 = 0; k0 < S; k0 += 64) {
    { // stage K [k][d] (async DMA when available) and V transposed [d][k]
      int row = tid >> 1;
      int dof = (tid & 1) * 32;
      const uint4* gk = (const uint4*)(Kb + (size_t)(k0 + row) * 64 + dof);
      uint4* dk = (uint4*)&sK[row * 64 + dof];
#if HAVE_ASYNC_COPY
      #pragma unroll
      for (int i = 0; i < 4; ++i) async_copy_b128(gk + i, dk + i);
#else
      #pragma unroll
      for (int i = 0; i < 4; ++i) dk[i] = gk[i];
#endif
      const uint4* gv = (const uint4*)(Vb + (size_t)(k0 + row) * 64 + dof);
      #pragma unroll
      for (int i = 0; i < 4; ++i) {
        union { uint4 q; unsigned short s[8]; } t;
        t.q = gv[i];
        #pragma unroll
        for (int jj = 0; jj < 8; ++jj)
          sV[(dof + 8*i + jj) * 64 + row] = t.s[jj];
      }
      // prefetch next k-block's dist rows (hides the 33.5MB index stream)
      if (k0 + 64 < S) {
        __builtin_prefetch(distb + (size_t)(q0 + (tid & 63)) * S + (k0 + 64), 0, 1);
        __builtin_prefetch(distb + (size_t)(q0 + (tid & 63)) * S + (k0 + 96), 0, 1);
      }
#if HAVE_ASYNC_COPY
      async_wait();
#endif
    }
    __syncthreads();

    // scores: 16q x 64k, Dh contraction in two K=32 steps
    v8f sc[4] = {};
    #pragma unroll
    for (int ks = 0; ks < 2; ++ks) {
      Frag16 aq = load_frag(sQ, qw, 64, ks * 32, lane);
      #pragma unroll
      for (int nt = 0; nt < 4; ++nt) {
        Frag16 bk = load_frag(sK, nt * 16, 64, ks * 32, lane);
        sc[nt] = wmma_bf16(aq, bk, sc[nt]);
      }
    }

    // scale + fused relative-bias gather (dist<0 => bias 0, not masked)
    float sv_[4][8];
    #pragma unroll
    for (int nt = 0; nt < 4; ++nt) {
      int kg = k0 + nt * 16 + (lane & 15);
      #pragma unroll
      for (int r = 0; r < 8; ++r) {
        int qg = q0 + qw + r + rofs;
        int d  = distb[(size_t)qg * S + kg];
        float bv = (d < 0) ? 0.f : bias_table[d * NUM_HEADS + h];
        sv_[nt][r] = sc[nt][r] * 0.125f + bv;   // 1/sqrt(64)
      }
    }

    // online softmax: rows live in fixed 16-lane groups (xor<=8 stays in group)
    #pragma unroll
    for (int r = 0; r < 8; ++r) {
      float m = fmaxf(fmaxf(sv_[0][r], sv_[1][r]), fmaxf(sv_[2][r], sv_[3][r]));
      #pragma unroll
      for (int off = 8; off > 0; off >>= 1)
        m = fmaxf(m, __shfl_xor(m, off, 32));
      float mnew  = fmaxf(mrow[r], m);
      float alpha = __expf(mrow[r] - mnew);
      mrow[r] = mnew;
      lrow[r] *= alpha;
      #pragma unroll
      for (int nt = 0; nt < 4; ++nt) o[nt][r] *= alpha;
      float ps = 0.f;
      #pragma unroll
      for (int nt = 0; nt < 4; ++nt) {
        float p = __expf(sv_[nt][r] - mnew);
        ps += p;
        sP[wave][(r + rofs) * 64 + nt * 16 + (lane & 15)] = f2bf(p);
      }
      #pragma unroll
      for (int off = 8; off > 0; off >>= 1)
        ps += __shfl_xor(ps, off, 32);
      lrow[r] += ps;
    }

    // O += P * V  (keys contraction in two K=32 steps; same-wave LDS in-order)
    #pragma unroll
    for (int ks = 0; ks < 2; ++ks) {
      Frag16 ap = load_frag(&sP[wave][0], 0, 64, ks * 32, lane);
      #pragma unroll
      for (int nt = 0; nt < 4; ++nt) {
        Frag16 bv = load_frag(sV, nt * 16, 64, ks * 32, lane);
        o[nt] = wmma_bf16(ap, bv, o[nt]);
      }
    }
    __syncthreads();
  }

  // normalize and write bf16 [B, S, H*Dh] for the output projection GEMM
  #pragma unroll
  for (int nt = 0; nt < 4; ++nt) {
    int dd = nt * 16 + (lane & 15);
    #pragma unroll
    for (int r = 0; r < 8; ++r) {
      int qg = q0 + qw + r + rofs;
      O[((size_t)b * S + qg) * D_MODEL + h * D_HEAD + dd] = f2bf(o[nt][r] / lrow[r]);
    }
  }
}

// ---------------------------------------------------------------------------
extern "C" void kernel_launch(void* const* d_in, const int* in_sizes, int n_in,
                              void* d_out, int out_size, void* d_ws, size_t ws_size,
                              hipStream_t stream) {
  const float* x          = (const float*)d_in[0];
  const int*   dist       = (const int*)d_in[1];
  const float* Wq         = (const float*)d_in[2];
  const float* bq         = (const float*)d_in[3];
  const float* Wk         = (const float*)d_in[4];
  const float* bk         = (const float*)d_in[5];
  const float* Wv         = (const float*)d_in[6];
  const float* bv         = (const float*)d_in[7];
  const float* Wo         = (const float*)d_in[8];
  const float* bo         = (const float*)d_in[9];
  const float* bias_table = (const float*)d_in[10];

  const long long BS = (long long)in_sizes[0] / D_MODEL;   // B*S
  const int S = (int)((long long)in_sizes[1] / BS);
  const int B = (int)(BS / S);
  const int M = (int)BS;

  unsigned short* Qb = (unsigned short*)d_ws;          // bf16 workspace
  unsigned short* Kb = Qb + (size_t)M * D_MODEL;
  unsigned short* Vb = Kb + (size_t)M * D_MODEL;
  unsigned short* Ob = Vb + (size_t)M * D_MODEL;

  dim3 gG(M / 64, D_MODEL / 64);
  gemm_wmma<0,1><<<gG, 128, 0, stream>>>(x, Wq, bq, Qb, M, D_MODEL, D_MODEL, S);
  gemm_wmma<0,1><<<gG, 128, 0, stream>>>(x, Wk, bk, Kb, M, D_MODEL, D_MODEL, S);
  gemm_wmma<0,1><<<gG, 128, 0, stream>>>(x, Wv, bv, Vb, M, D_MODEL, D_MODEL, S);

  const int BH = B * NUM_HEADS;
  long long npair = (long long)BH * S * (D_HEAD / 2);
  rope_kernel<<<(unsigned)((npair + 255) / 256), 256, 0, stream>>>(Qb, Kb, npair, S);

  attn_wmma<<<dim3(S / 64, BH), 128, 0, stream>>>(Qb, Kb, Vb, dist, bias_table, Ob, S);

  gemm_wmma<1,0><<<gG, 128, 0, stream>>>(Ob, Wo, bo, (float*)d_out, M, D_MODEL, D_MODEL, S);
}
